// CompressedLinear_17575006175590
// MI455X (gfx1250) — matrix operand order repaired and checked by reading, not compiled, not run on hardware
//
#include <hip/hip_runtime.h>

// CompressedLinear on gfx1250:  out = (x_f16 · w_codes_f16^T) * scale + bias
// M=4096 (B*S), N=11008 (DOUT), K=4096 (DIN). f16 WMMA, f32 accumulate.
//
// Roofline: 369 GFLOP, ~424 MB min HBM traffic (18 us @ 23.3 TB/s) vs ~74 us
// at dense-f16 matrix peak -> compute-bound -> f16 WMMA path. int8 weight
// codes are exact in f16; only x f32->f16 is lossy. 256x256 tiles keep
// L2 tile traffic at ~5.8 GB with the 180 MB weight matrix L2-resident.

typedef _Float16 half_t;
typedef __attribute__((ext_vector_type(16))) _Float16 v16h;
typedef __attribute__((ext_vector_type(8)))  _Float16 v8h;
typedef __attribute__((ext_vector_type(4)))  _Float16 v4h;
typedef __attribute__((ext_vector_type(8)))  float    v8f;
typedef __attribute__((ext_vector_type(4)))  float    v4f;
typedef __attribute__((ext_vector_type(4)))  int      v4i;

#define M_DIM 4096      // B*S
#define N_DIM 11008     // DOUT
#define K_DIM 4096      // DIN

#define BM 256
#define BN 256
#define BK 32
#define LDP (BK + 8)    // padded LDS row stride in halves (80B rows -> no bank conflicts)
#define NTHREADS 512    // 16 waves (wave32)
#define NKT (K_DIM / BK)

__global__ __launch_bounds__(NTHREADS)
void compressed_linear_wmma(const float* __restrict__ x,
                            const int*   __restrict__ w,
                            const float* __restrict__ scale_ptr,
                            const float* __restrict__ bias,
                            float*       __restrict__ out)
{
    // single-buffered LDS tiles (40 KB total), register-level K prefetch instead
    __shared__ half_t lA[BM * LDP];   // [m][k] f16
    __shared__ half_t lB[BN * LDP];   // [n][k] f16 (weights, K-major per row)

    const int tid  = threadIdx.x;
    const int lane = tid & 31;
    const int wid  = tid >> 5;                 // 0..15

    const int bm = blockIdx.y * BM;
    const int bn = blockIdx.x * BN;

    // 4x4 wave grid; each wave computes a 64x64 sub-tile = 4x4 WMMA tiles
    const int wm = (wid & 3) * 64;
    const int wn = (wid >> 2) * 64;

    // Uniform bases + 32-bit per-lane byte offsets; K-tile advance is uniform.
    const char* xbase = (const char*)(x + (size_t)bm * K_DIM);
    const char* wbase = (const char*)(w + (size_t)bn * K_DIM);

    unsigned int off[4];
    int ldsOff[4];
    #pragma unroll
    for (int i = 0; i < 4; ++i) {
        const int idx = tid + i * NTHREADS;    // 0..2047
        const int row = idx >> 3;              // 0..255
        const int kc  = (idx & 7) << 2;        // 0,4,..,28
        off[i]    = (unsigned int)((row * K_DIM + kc) * 4);   // bytes
        ldsOff[i] = row * LDP + kc;                           // halves
    }

    v8f acc[4][4];
    #pragma unroll
    for (int i = 0; i < 4; ++i)
        #pragma unroll
        for (int j = 0; j < 4; ++j) {
            v8f z = {0.f, 0.f, 0.f, 0.f, 0.f, 0.f, 0.f, 0.f};
            acc[i][j] = z;
        }

    v4f aReg[4];
    v4i bReg[4];

    // ---------------- prologue: fetch K-tile 0 and commit to LDS ----------------
    #pragma unroll
    for (int i = 0; i < 4; ++i) {
        aReg[i] = *(const v4f*)(xbase + (size_t)off[i]);
        bReg[i] = *(const v4i*)(wbase + (size_t)off[i]);
    }
    #pragma unroll
    for (int i = 0; i < 4; ++i) {
        v4h pa = { (half_t)aReg[i].x, (half_t)aReg[i].y,
                   (half_t)aReg[i].z, (half_t)aReg[i].w };
        *(v4h*)&lA[ldsOff[i]] = pa;
        v4h pb = { (half_t)(float)bReg[i].x, (half_t)(float)bReg[i].y,
                   (half_t)(float)bReg[i].z, (half_t)(float)bReg[i].w };
        *(v4h*)&lB[ldsOff[i]] = pb;
    }
    __syncthreads();

    // ---------------- main K loop ----------------
    for (int kt = 0; kt < NKT; ++kt) {
        // prefetch next K-tile into registers; VMEM latency hides under the WMMAs
        if (kt + 1 < NKT) {
            const unsigned int kByte = (unsigned int)((kt + 1) * BK * 4);
            #pragma unroll
            for (int i = 0; i < 4; ++i) {
                aReg[i] = *(const v4f*)(xbase + (size_t)(off[i] + kByte));
                bReg[i] = *(const v4i*)(wbase + (size_t)(off[i] + kByte));
            }
        }

        // Preload ALL fragments (4 A + 4 B) into dedicated registers so a
        // single dscnt wait covers the full 16-WMMA burst and no octet is
        // reused mid-burst (fewer WMMA hazard NOPs).
        //
        // A frag (16x32 f16): row = lane%16, kbase = 8*(lane/16);
        //   halves [0..7] = K kbase.. ; halves [8..15] = K 16+kbase..
        // B frag (32x16 f16): col = lane%16, kbase = 8*(lane/16); same K split as A.
        v16h afrag[4];
        v16h bfrag[4];
        #pragma unroll
        for (int t = 0; t < 4; ++t) {
            const int row = wm + t * 16 + (lane & 15);
            const int kb  = (lane >> 4) * 8;
            const v8h lo = *(const v8h*)&lA[row * LDP + kb];
            const v8h hi = *(const v8h*)&lA[row * LDP + kb + 16];
            afrag[t] = __builtin_shufflevector(lo, hi,
                0,1,2,3,4,5,6,7,8,9,10,11,12,13,14,15);
        }
        #pragma unroll
        for (int t = 0; t < 4; ++t) {
            const int col = wn + t * 16 + (lane & 15);
            const int kb  = (lane >> 4) * 8;
            const v8h lo = *(const v8h*)&lB[col * LDP + kb];
            const v8h hi = *(const v8h*)&lB[col * LDP + kb + 16];
            bfrag[t] = __builtin_shufflevector(lo, hi,
                0,1,2,3,4,5,6,7,8,9,10,11,12,13,14,15);
        }

        #pragma unroll
        for (int bt = 0; bt < 4; ++bt)
            #pragma unroll
            for (int am = 0; am < 4; ++am)
                acc[am][bt] = __builtin_amdgcn_wmma_f32_16x16x32_f16(
                    /*neg_a=*/false, afrag[am], /*neg_b=*/false, bfrag[bt],
                    /*c_mod=*/(short)0, acc[am][bt],
                    /*reuse_a=*/false, /*reuse_b=*/false);

        __syncthreads();   // everyone done reading this K-tile

        if (kt + 1 < NKT) {
            #pragma unroll
            for (int i = 0; i < 4; ++i) {
                v4h pa = { (half_t)aReg[i].x, (half_t)aReg[i].y,
                           (half_t)aReg[i].z, (half_t)aReg[i].w };
                *(v4h*)&lA[ldsOff[i]] = pa;
                v4h pb = { (half_t)(float)bReg[i].x, (half_t)(float)bReg[i].y,
                           (half_t)(float)bReg[i].z, (half_t)(float)bReg[i].w };
                *(v4h*)&lB[ldsOff[i]] = pb;
            }
            __syncthreads();   // new tile visible before next iteration's reads
        }
    }

    // ---------------- epilogue: out = scale*acc + bias ----------------
    // C/D layout: col = lane%16, rows = 8*(lane/16) + v (v = VGPR 0..7).
    // Non-temporal stores: output is write-once; keep L2 for the weight matrix.
    const float s = *scale_ptr;
    #pragma unroll
    for (int t = 0; t < 4; ++t) {
        const int col = bn + wn + t * 16 + (lane & 15);
        const float bv = bias[col];
        #pragma unroll
        for (int am = 0; am < 4; ++am) {
            const int rbase = bm + wm + am * 16 + (lane >> 4) * 8;
            #pragma unroll
            for (int v = 0; v < 8; ++v) {
                float* p = out + (size_t)(rbase + v) * N_DIM + col;
                __builtin_nontemporal_store(acc[am][t][v] * s + bv, p);
            }
        }
    }
}

extern "C" void kernel_launch(void* const* d_in, const int* in_sizes, int n_in,
                              void* d_out, int out_size, void* d_ws, size_t ws_size,
                              hipStream_t stream) {
    (void)in_sizes; (void)n_in; (void)out_size; (void)d_ws; (void)ws_size;
    const float* x     = (const float*)d_in[0];
    const int*   w     = (const int*)  d_in[1];
    const float* scale = (const float*)d_in[2];
    const float* bias  = (const float*)d_in[3];
    float*       out   = (float*)d_out;

    dim3 grid(N_DIM / BN, M_DIM / BM);   // (43, 16)
    dim3 block(NTHREADS);                // 512 threads = 16 wave32 waves
    compressed_linear_wmma<<<grid, block, 0, stream>>>(x, w, scale, bias, out);
}
